// S6_49847390437668
// MI455X (gfx1250) — matrix operand order repaired and checked by reading, not compiled, not run on hardware
//
#include <hip/hip_runtime.h>
#include <hip/hip_bf16.h>

// CDNA5 / gfx1250: wave32, WMMA 16x16x32 bf16 -> f32 accumulate.
typedef __attribute__((ext_vector_type(16))) __bf16 v16bf;
typedef __attribute__((ext_vector_type(8)))  float  v8f;
typedef __attribute__((ext_vector_type(4)))  float  v4f;

#define DMODEL 64
#define NSTATE 64
#define TOK_PER_WAVE 16
#define WAVES_PER_BLK 4
#define THREADS_PER_BLK (WAVES_PER_BLK * 32)          // 128
#define TOK_PER_BLK (TOK_PER_WAVE * WAVES_PER_BLK)    // 64

// y[t,d] = x[t,d] * softplus((x@W1+b1)[t,d]) * sum_n (x@W2+b2)[t,n]*(x@W3+b3)[t,n]
__global__ __launch_bounds__(THREADS_PER_BLK, 4)
void s6_fused_wmma_kernel(const float* __restrict__ x,
                          const float* __restrict__ W1, const float* __restrict__ b1,
                          const float* __restrict__ W2, const float* __restrict__ b2,
                          const float* __restrict__ W3, const float* __restrict__ b3,
                          float* __restrict__ y)
{
    // Transposed bf16 weights in LDS: ldsW[g][n][k], contiguous in k so each
    // lane's B-fragment (16 bf16 = 32B) is two ds_load_b128.
    __shared__ __align__(32) __bf16 ldsW[3 * DMODEL * NSTATE];
    __shared__ float ldsB[3 * NSTATE];

    const int tid = threadIdx.x;

    // ---- Cooperative weight stage: f32 [k][n] global -> bf16 [n][k] LDS ----
    #pragma unroll
    for (int g = 0; g < 3; ++g) {
        const float* W = (g == 0) ? W1 : (g == 1) ? W2 : W3;
        for (int i = tid; i < DMODEL * NSTATE; i += THREADS_PER_BLK) {
            const int k = i >> 6;        // row of W (input feature)
            const int n = i & 63;        // col of W (output feature)
            ldsW[g * 4096 + n * 64 + k] = (__bf16)W[i];
        }
    }
    if (tid < NSTATE) {
        ldsB[tid]       = b1[tid];
        ldsB[64 + tid]  = b2[tid];
        ldsB[128 + tid] = b3[tid];
    }
    __syncthreads();

    const int wave = tid >> 5;
    const int lane = tid & 31;
    const int half = lane >> 4;   // which 16-lane half of the wave
    const int l16  = lane & 15;
    const int tileBase = blockIdx.x * TOK_PER_BLK + wave * TOK_PER_WAVE;

    // 12 accumulator tiles: [0..3]=xW1 n-tiles, [4..7]=xW2, [8..11]=xW3
    v8f acc[12];
    #pragma unroll
    for (int i = 0; i < 12; ++i) acc[i] = (v8f){0.f,0.f,0.f,0.f,0.f,0.f,0.f,0.f};

    // A-fragment source row for this lane (ISA: lane L holds matrix row L%16)
    const float* xrowA = x + (size_t)(tileBase + l16) * DMODEL;
    // Per-lane B-fragment base inside one weight matrix: n = nt*16+l16 row,
    // k offset = ks*32 + half*16 (16 contiguous bf16 = 32 bytes).
    const __bf16* wbase = &ldsW[(size_t)l16 * 64 + half * 16];

    #pragma unroll
    for (int ks = 0; ks < 2; ++ks) {          // K = 64 in two steps of 32
        // ---- Build A fragment in contiguous k order (same as B) ----
        // halfs 0..15 : K = ks*32 + half*16 + (0..15)
        const int kb0 = ks * 32 + half * 16;
        const int kb1 = kb0 + 8;
        const v4f f0 = *(const v4f*)(xrowA + kb0);
        const v4f f1 = *(const v4f*)(xrowA + kb0 + 4);
        const v4f f2 = *(const v4f*)(xrowA + kb1);
        const v4f f3 = *(const v4f*)(xrowA + kb1 + 4);
        v16bf a;
        #pragma unroll
        for (int i = 0; i < 4; ++i) {
            a[i]      = (__bf16)f0[i];
            a[4 + i]  = (__bf16)f1[i];
            a[8 + i]  = (__bf16)f2[i];
            a[12 + i] = (__bf16)f3[i];
        }

        #pragma unroll
        for (int g = 0; g < 3; ++g) {
            // Batch-load all four n-tile B fragments into distinct registers
            // so 8 ds_load_b128 are in flight before the first WMMA consumes
            // one (avoids a full s_wait_dscnt 0 per WMMA).
            const __bf16* wg = wbase + g * 4096 + ks * 32;
            const v16bf bf0 = *(const v16bf*)(wg + 0 * 16 * 64);
            const v16bf bf1 = *(const v16bf*)(wg + 1 * 16 * 64);
            const v16bf bf2 = *(const v16bf*)(wg + 2 * 16 * 64);
            const v16bf bf3 = *(const v16bf*)(wg + 3 * 16 * 64);
            acc[g * 4 + 0] = __builtin_amdgcn_wmma_f32_16x16x32_bf16(
                false, a, false, bf0, (short)0, acc[g * 4 + 0], false, false);
            acc[g * 4 + 1] = __builtin_amdgcn_wmma_f32_16x16x32_bf16(
                false, a, false, bf1, (short)0, acc[g * 4 + 1], false, false);
            acc[g * 4 + 2] = __builtin_amdgcn_wmma_f32_16x16x32_bf16(
                false, a, false, bf2, (short)0, acc[g * 4 + 2], false, false);
            acc[g * 4 + 3] = __builtin_amdgcn_wmma_f32_16x16x32_bf16(
                false, a, false, bf3, (short)0, acc[g * 4 + 3], false, false);
        }
    }

    // ---- Per-lane bias values for each n-tile ----
    float b1v[4], b2v[4], b3v[4];
    #pragma unroll
    for (int nt = 0; nt < 4; ++nt) {
        b1v[nt] = ldsB[nt * 16 + l16];
        b2v[nt] = ldsB[64 + nt * 16 + l16];
        b3v[nt] = ldsB[128 + nt * 16 + l16];
    }

    // ---- Epilogue: D layout is VGPR r -> row (r or r+8), lane%16 -> column ----
    #pragma unroll
    for (int r = 0; r < 8; ++r) {
        // s = sum_n Bm*C for this row: per-lane partial over the 4 n-tiles,
        // then butterfly-reduce across the 16-lane half (token r vs r+8).
        float p = 0.f;
        #pragma unroll
        for (int nt = 0; nt < 4; ++nt) {
            const float bm = acc[4 + nt][r] + b2v[nt];
            const float cv = acc[8 + nt][r] + b3v[nt];
            p = fmaf(bm, cv, p);
        }
        p += __shfl_xor(p, 1, 16);
        p += __shfl_xor(p, 2, 16);
        p += __shfl_xor(p, 4, 16);
        p += __shfl_xor(p, 8, 16);

        const int token = tileBase + half * 8 + r;
        const float* xr = x + (size_t)token * DMODEL;
        float* yr = y + (size_t)token * DMODEL;
        #pragma unroll
        for (int nt = 0; nt < 4; ++nt) {
            const int d = nt * 16 + l16;
            const float z  = acc[nt][r] + b1v[nt];
            // numerically stable softplus
            const float sp = fmaxf(z, 0.f) + log1pf(expf(-fabsf(z)));
            yr[d] = xr[d] * sp * p;
        }
    }
}

extern "C" void kernel_launch(void* const* d_in, const int* in_sizes, int n_in,
                              void* d_out, int out_size, void* d_ws, size_t ws_size,
                              hipStream_t stream) {
    (void)n_in; (void)out_size; (void)d_ws; (void)ws_size;
    const float* x  = (const float*)d_in[0];
    const float* W1 = (const float*)d_in[1];
    const float* b1 = (const float*)d_in[2];
    const float* W2 = (const float*)d_in[3];
    const float* b2 = (const float*)d_in[4];
    const float* W3 = (const float*)d_in[5];
    const float* b3 = (const float*)d_in[6];
    // d_in[7] (A) is dead: dA is multiplied by zero in the reference.
    float* y = (float*)d_out;

    const int tokens = in_sizes[0] / DMODEL;        // BATCH*SEQ = 8192
    const int blocks = tokens / TOK_PER_BLK;        // 128
    s6_fused_wmma_kernel<<<blocks, THREADS_PER_BLK, 0, stream>>>(
        x, W1, b1, W2, b2, W3, b3, y);
}
